// SliceModel_4372276707961
// MI455X (gfx1250) — compile-verified, matrix operand order checked by reading, not run on hardware
//
#include <hip/hip_runtime.h>

// SliceLSTM fused kernel for gfx1250 (MI455X) — register-lean, branch-free inner loop.
//   x:[4096,512,50] f32 -> out:[4096,1] f32
// One wave = 16 batch rows. Per timestep:
//   - transpose current 16x32 gate tile through LDS (ds_store_2addr / ds_load_b128)
//   - 2 lanes/batch run the serial LSTM recurrences (rcp-based activations)
//   - load next row (prefetched 8 rows ahead), pack bf16, 4x v_wmma_f32_16x16x32_bf16

#define BATCH 4096
#define TSEQ  512
#define DIN   50

typedef __attribute__((ext_vector_type(16))) __bf16 v16bf;
typedef __attribute__((ext_vector_type(8)))  float  v8f;

union BfPack {
  v16bf v;
  unsigned int u[8];
};

// truncating f32->bf16 pack of two values into one dword
__device__ __forceinline__ unsigned int pack2t(float lo, float hi) {
  return (__float_as_uint(hi) & 0xffff0000u) | (__float_as_uint(lo) >> 16);
}
__device__ __forceinline__ float sigm(float x) {
  return __builtin_amdgcn_rcpf(1.0f + __expf(-x));
}
__device__ __forceinline__ float tanhx(float x) {
  float e = __expf(-2.0f * fabsf(x));
  float r = (1.0f - e) * __builtin_amdgcn_rcpf(1.0f + e);
  return copysignf(r, x);
}

// one serial step of both stacked per-lane LSTMs (all indices constant after inlining)
__device__ __forceinline__ void lstm_step(const float* __restrict__ g,
                                          const float* __restrict__ Whh,
                                          const float* __restrict__ W2I,
                                          const float* __restrict__ W2H,
                                          const float* __restrict__ Bb2,
                                          float* __restrict__ h1, float* __restrict__ c1v,
                                          float* __restrict__ h2, float* __restrict__ c2v) {
  // layer-1 LSTM (H=4), gate order i,f,g,o; old h1 for all dots
  float h1n[4];
#pragma unroll
  for (int j = 0; j < 4; ++j) {
    float pi = g[j], pf = g[4 + j], pg = g[8 + j], po = g[12 + j];
#pragma unroll
    for (int k = 0; k < 4; ++k) {
      pi += Whh[j * 4 + k]        * h1[k];
      pf += Whh[(4 + j) * 4 + k]  * h1[k];
      pg += Whh[(8 + j) * 4 + k]  * h1[k];
      po += Whh[(12 + j) * 4 + k] * h1[k];
    }
    c1v[j] = sigm(pf) * c1v[j] + sigm(pi) * tanhx(pg);
    h1n[j] = sigm(po) * tanhx(c1v[j]);
  }
#pragma unroll
  for (int j = 0; j < 4; ++j) h1[j] = h1n[j];

  // layer-2 LSTM (H=2); this lane's slice consumes exactly its own h1
  float h2n[2];
#pragma unroll
  for (int j = 0; j < 2; ++j) {
    float pi = Bb2[j], pf = Bb2[2 + j], pg = Bb2[4 + j], po = Bb2[6 + j];
#pragma unroll
    for (int k = 0; k < 4; ++k) {
      pi += W2I[j * 4 + k]       * h1[k];
      pf += W2I[(2 + j) * 4 + k] * h1[k];
      pg += W2I[(4 + j) * 4 + k] * h1[k];
      po += W2I[(6 + j) * 4 + k] * h1[k];
    }
#pragma unroll
    for (int k = 0; k < 2; ++k) {
      pi += W2H[j * 2 + k]       * h2[k];
      pf += W2H[(2 + j) * 2 + k] * h2[k];
      pg += W2H[(4 + j) * 2 + k] * h2[k];
      po += W2H[(6 + j) * 2 + k] * h2[k];
    }
    c2v[j] = sigm(pf) * c2v[j] + sigm(pi) * tanhx(pg);
    h2n[j] = sigm(po) * tanhx(c2v[j]);
  }
  h2[0] = h2n[0]; h2[1] = h2n[1];
}

__global__ __launch_bounds__(256)
void slicelstm_fused(const float* __restrict__ x,
                     const float* __restrict__ w1a_ih, const float* __restrict__ w1a_hh,
                     const float* __restrict__ b1a,
                     const float* __restrict__ w1b_ih, const float* __restrict__ w1b_hh,
                     const float* __restrict__ b1b,
                     const float* __restrict__ w2a_ih, const float* __restrict__ w2a_hh,
                     const float* __restrict__ b2a,
                     const float* __restrict__ w2b_ih, const float* __restrict__ w2b_hh,
                     const float* __restrict__ b2b,
                     const float* __restrict__ w_out, const float* __restrict__ b_out,
                     float* __restrict__ out) {
  const int lane = threadIdx.x & 31;
  const int wid  = threadIdx.x >> 5;
  const int tile = blockIdx.x * (blockDim.x >> 5) + wid;   // 0..255
  const int b0   = tile * 16;
  const int nloc = lane & 15;   // N column (gate) within tile / A row selector
  const int hi   = lane >> 4;   // K-half selector per WMMA operand layout

  __shared__ float lds[8][16][32];   // per-wave 16x32 gate transpose buffer

  // ---- B operands: padded block-diagonal weight Wp[32][64], bf16 ----
  BfPack Bop[2][2];   // [gate-tile ng][K-chunk kc]
#pragma unroll
  for (int ng = 0; ng < 2; ++ng) {
#pragma unroll
    for (int kc = 0; kc < 2; ++kc) {
      const int n = ng * 16 + nloc;
#pragma unroll
      for (int j = 0; j < 16; j += 2) {
        float wv[2];
#pragma unroll
        for (int q = 0; q < 2; ++q) {
          const int k = kc * 32 + hi * 16 + j + q;
          float v = 0.0f;
          if (n < 16) {
            if (k < 25) v = w1a_ih[n * 25 + k];
          } else {
            const int ks = k - 25;
            if (ks >= 0 && ks < 25) v = w1b_ih[(n - 16) * 25 + ks];
          }
          wv[q] = v;
        }
        Bop[ng][kc].u[j >> 1] = pack2t(wv[0], wv[1]);
      }
    }
  }

  // bias scalars (broadcast into accumulators at each init; C layout: N = lane%16)
  const float ba = b1a[nloc];
  const float bb = b1b[nloc];

  // ---- per-lane recurrence weights (even lane = slice a, odd = slice b) ----
  const bool odd = (lane & 1) != 0;
  const float* whhG = odd ? w1b_hh : w1a_hh;   // [16][4]
  const float* w2iG = odd ? w2b_ih : w2a_ih;   // [8][4]
  const float* w2hG = odd ? w2b_hh : w2a_hh;   // [8][2]
  const float* b2G  = odd ? b2b    : b2a;      // [8]
  float Whh[64], W2I[32], W2H[16], Bb2[8];
#pragma unroll
  for (int i = 0; i < 64; ++i) Whh[i] = whhG[i];
#pragma unroll
  for (int i = 0; i < 32; ++i) W2I[i] = w2iG[i];
#pragma unroll
  for (int i = 0; i < 16; ++i) W2H[i] = w2hG[i];
#pragma unroll
  for (int i = 0; i < 8;  ++i) Bb2[i] = b2G[i];

  // ---- state ----
  float h1[4] = {0.f, 0.f, 0.f, 0.f}, c1v[4] = {0.f, 0.f, 0.f, 0.f};
  float h2[2] = {0.f, 0.f},           c2v[2] = {0.f, 0.f};

  // per-lane addressing (closed-form, branch-free zero padding for the K tail):
  //   chunk0: k = hi*16 + 2j          (always in-bounds, j=0..7)
  //   chunk1: k = (hi?48:32) + (hi?0:2)*j ; pairs j>=1 of hi-lanes are masked to 0
  const float* xrow = x + (size_t)(b0 + nloc) * TSEQ * DIN;
  const int kA   = hi * 16;
  const int kB   = hi ? 48 : 32;
  const int st1  = hi ? 0 : 2;
  const unsigned mskT = hi ? 0u : 0xffffffffu;

  // ---- prologue: load+pack A(0), compute gate tile d(0) ----
  v8f d0, d1;
  {
    const float* bp0 = xrow + kA;
    const float* bp1 = xrow + kB;
    BfPack a0, a1;
#pragma unroll
    for (int j = 0; j < 8; ++j) {
      const float2 f0 = *(const float2*)(bp0 + 2 * j);
      const float2 f1 = *(const float2*)(bp1 + st1 * j);
      a0.u[j] = pack2t(f0.x, f0.y);
      unsigned u = pack2t(f1.x, f1.y);
      if (j) u &= mskT;
      a1.u[j] = u;
    }
#pragma unroll
    for (int r = 0; r < 8; ++r) { d0[r] = ba; d1[r] = bb; }
    d0 = __builtin_amdgcn_wmma_f32_16x16x32_bf16(false, a0.v, false, Bop[0][0].v,
                                                 (short)0, d0, false, false);
    d0 = __builtin_amdgcn_wmma_f32_16x16x32_bf16(false, a1.v, false, Bop[0][1].v,
                                                 (short)0, d0, false, false);
    d1 = __builtin_amdgcn_wmma_f32_16x16x32_bf16(false, a0.v, false, Bop[1][0].v,
                                                 (short)0, d1, false, false);
    d1 = __builtin_amdgcn_wmma_f32_16x16x32_bf16(false, a1.v, false, Bop[1][1].v,
                                                 (short)0, d1, false, false);
  }

  float g[16];
  const float* gp = &lds[wid][lane >> 1][(lane & 1) * 16];
  const float* nxt = xrow + DIN;   // row for t+1, advanced by DIN each iteration

  // ---- steady state: 511 iterations ----
  for (int t = 0; t < TSEQ - 1; ++t) {
    __builtin_prefetch(nxt + 8 * DIN, 0, 1);   // global_prefetch_b8, 8 rows ahead

    // (1) D layout -> LDS: element (M=r+8*hi, N=nloc[+16]) per lane
#pragma unroll
    for (int r = 0; r < 8; ++r) {
      const int m = r + 8 * hi;
      lds[wid][m][nloc]      = d0[r];
      lds[wid][m][16 + nloc] = d1[r];
    }
    __builtin_amdgcn_wave_barrier();   // intra-wave LDS is in-order; block reordering

    // (2) lane 2m+s reads the 16 gates of batch m, slice s
#pragma unroll
    for (int j = 0; j < 16; ++j) g[j] = gp[j];
    __builtin_amdgcn_wave_barrier();

    // (3) serial recurrences for t
    lstm_step(g, Whh, W2I, W2H, Bb2, h1, c1v, h2, c2v);

    // (4) load (prefetched) row t+1, pack bf16, launch next gate-tile WMMAs
    {
      const float* bp0 = nxt + kA;
      const float* bp1 = nxt + kB;
      BfPack a0, a1;
#pragma unroll
      for (int j = 0; j < 8; ++j) {
        const float2 f0 = *(const float2*)(bp0 + 2 * j);
        const float2 f1 = *(const float2*)(bp1 + st1 * j);
        a0.u[j] = pack2t(f0.x, f0.y);
        unsigned u = pack2t(f1.x, f1.y);
        if (j) u &= mskT;
        a1.u[j] = u;
      }
#pragma unroll
      for (int r = 0; r < 8; ++r) { d0[r] = ba; d1[r] = bb; }
      d0 = __builtin_amdgcn_wmma_f32_16x16x32_bf16(false, a0.v, false, Bop[0][0].v,
                                                   (short)0, d0, false, false);
      d0 = __builtin_amdgcn_wmma_f32_16x16x32_bf16(false, a1.v, false, Bop[0][1].v,
                                                   (short)0, d0, false, false);
      d1 = __builtin_amdgcn_wmma_f32_16x16x32_bf16(false, a0.v, false, Bop[1][0].v,
                                                   (short)0, d1, false, false);
      d1 = __builtin_amdgcn_wmma_f32_16x16x32_bf16(false, a1.v, false, Bop[1][1].v,
                                                   (short)0, d1, false, false);
    }
    nxt += DIN;   // strength-reduced row pointer (1 v_add_nc_u64)
  }

  // ---- epilogue: recurrence for the final timestep ----
#pragma unroll
  for (int r = 0; r < 8; ++r) {
    const int m = r + 8 * hi;
    lds[wid][m][nloc]      = d0[r];
    lds[wid][m][16 + nloc] = d1[r];
  }
  __builtin_amdgcn_wave_barrier();
#pragma unroll
  for (int j = 0; j < 16; ++j) g[j] = gp[j];
  __builtin_amdgcn_wave_barrier();
  lstm_step(g, Whh, W2I, W2H, Bb2, h1, c1v, h2, c2v);

  // output head: h=[h2a,h2b], out = sigmoid(h . w_out + b_out)
  float pp = odd ? (h2[0] * w_out[2] + h2[1] * w_out[3])
                 : (h2[0] * w_out[0] + h2[1] * w_out[1]);
  pp += __shfl_xor(pp, 1, 32);
  if (!odd) {
    out[b0 + (lane >> 1)] = sigm(pp + b_out[0]);
  }
}

extern "C" void kernel_launch(void* const* d_in, const int* in_sizes, int n_in,
                              void* d_out, int out_size, void* d_ws, size_t ws_size,
                              hipStream_t stream) {
  (void)in_sizes; (void)n_in; (void)out_size; (void)d_ws; (void)ws_size;
  const float* x      = (const float*)d_in[0];
  const float* w1a_ih = (const float*)d_in[1];
  const float* w1a_hh = (const float*)d_in[2];
  const float* b1a    = (const float*)d_in[3];
  const float* w1b_ih = (const float*)d_in[4];
  const float* w1b_hh = (const float*)d_in[5];
  const float* b1b    = (const float*)d_in[6];
  const float* w2a_ih = (const float*)d_in[7];
  const float* w2a_hh = (const float*)d_in[8];
  const float* b2a    = (const float*)d_in[9];
  const float* w2b_ih = (const float*)d_in[10];
  const float* w2b_hh = (const float*)d_in[11];
  const float* b2b    = (const float*)d_in[12];
  const float* w_out  = (const float*)d_in[13];
  const float* b_out  = (const float*)d_in[14];
  float* out = (float*)d_out;

  // 4096 batches / 16 per wave = 256 waves; 8 waves per block -> 32 blocks
  dim3 block(256), grid(BATCH / 16 / 8);
  slicelstm_fused<<<grid, block, 0, stream>>>(
      x, w1a_ih, w1a_hh, b1a, w1b_ih, w1b_hh, b1b,
      w2a_ih, w2a_hh, b2a, w2b_ih, w2b_hh, b2b, w_out, b_out, out);
}
